// OptimizedQuantizedLinear_62268435858096
// MI455X (gfx1250) — compile-verified, matrix operand order checked by reading, not run on hardware
//
#include <hip/hip_runtime.h>

// Group-quantized linear: out = x @ dequant(Wq, scales)^T + bias
// Phase 1: dequantize int8-in-int32 weights (group=64 scales) -> f16 in d_ws.
// Phase 2: f16 WMMA GEMM, x converted fp32->f16 on the fly, W tiles fetched
//          with CDNA5 async global->LDS copies (ASYNCcnt).
// Working set after prepass: x 128MB fp32 + W 32MB f16 = 160MB < 192MB L2.

typedef _Float16 v4h  __attribute__((ext_vector_type(4)));
typedef _Float16 v8h  __attribute__((ext_vector_type(8)));
typedef _Float16 v16h __attribute__((ext_vector_type(16)));
typedef float    v8f  __attribute__((ext_vector_type(8)));

#define D_IN    4096
#define D_OUT   4096
#define M_TOT   8192
#define BM      128
#define BN      128
#define BK      32
#define KSTEPS  (D_IN / BK)           // 128
#define ROWB    96                    // LDS bytes per tile row: 64B data + 32B pad
#define TILEB   (BM * ROWB)           // 12288 B per tile buffer
#define X_OFF   0                     // two X buffers at [0, 24576)
#define W_OFF   (2 * TILEB)           // two W buffers at [24576, 49152)
#define LDS_BYTES (4 * TILEB)         // 48 KB

// ---------------- Phase 1: one-shot weight dequant (int8 -> f16) -------------
__global__ __launch_bounds__(256)
void dequant_w_kernel(const int* __restrict__ wq,
                      const float* __restrict__ wscale,
                      _Float16* __restrict__ wh)
{
    // 8 consecutive elements per thread; 8-aligned chunk never crosses a
    // 64-element scale group.
    const size_t base = ((size_t)blockIdx.x * 256u + threadIdx.x) * 8u;
    const float s = wscale[base >> 6];
    const int4 a = *(const int4*)(wq + base);
    const int4 b = *(const int4*)(wq + base + 4);
    v8h h = { (_Float16)((float)a.x * s), (_Float16)((float)a.y * s),
              (_Float16)((float)a.z * s), (_Float16)((float)a.w * s),
              (_Float16)((float)b.x * s), (_Float16)((float)b.y * s),
              (_Float16)((float)b.z * s), (_Float16)((float)b.w * s) };
    *(v8h*)(wh + base) = h;
}

// ---------------- Phase 2: WMMA GEMM ----------------------------------------
__global__ __launch_bounds__(256)
void qlinear_wmma_kernel(const float* __restrict__ x,
                         const _Float16* __restrict__ wh,
                         const float* __restrict__ bias,
                         float* __restrict__ out)
{
    __shared__ __attribute__((aligned(128))) char lds[LDS_BYTES];

    const int tid   = threadIdx.x;
    const int lane  = tid & 31;           // wave32
    const int wave  = tid >> 5;           // 0..7
    const int waveM = (wave & 1) * 64;    // 2 waves along M
    const int waveN = (wave >> 1) * 32;   // 4 waves along N

    const int m0 = blockIdx.y * BM;
    const int n0 = blockIdx.x * BN;

    // ---- X tile: global float4 -> cvt f16 -> LDS (needs conversion) ----
    const int row_ = tid >> 3;            // 0..31, +32 per i
    const int kc_  = (tid & 7) * 4;       // f16 element offset in row (0..28)
    float4 xr[4];

    auto loadRegsX = [&](int k0) {
        #pragma unroll
        for (int i = 0; i < 4; ++i) {
            const int row = row_ + i * 32;
            xr[i] = *(const float4*)(x + (size_t)(m0 + row) * D_IN + k0 + kc_);
        }
        if (k0 + 2 * BK <= D_IN - BK)
            __builtin_prefetch(x + (size_t)(m0 + row_) * D_IN + k0 + 2 * BK + kc_, 0, 1);
    };

    auto storeLdsX = [&](int buf) {
        #pragma unroll
        for (int i = 0; i < 4; ++i) {
            const int row = row_ + i * 32;
            v4h hx = { (_Float16)xr[i].x, (_Float16)xr[i].y,
                       (_Float16)xr[i].z, (_Float16)xr[i].w };
            *(v4h*)(lds + X_OFF + buf * TILEB + row * ROWB + kc_ * 2) = hx;
        }
    };

    // ---- W tile: pre-converted f16 -> CDNA5 async copy straight to LDS ----
    // 128 rows x 64B = 512 16B-chunks, 256 threads -> 2 chunks each.
    auto issueAsyncW = [&](int buf, int k0) {
        #pragma unroll
        for (int i = 0; i < 2; ++i) {
            const int idx = tid + i * 256;
            const int row = idx >> 2;
            const int c   = (idx & 3) * 16;   // byte chunk within row
            const unsigned ldsa = (unsigned)(W_OFF + buf * TILEB + row * ROWB + c);
            const unsigned long long ga =
                (unsigned long long)(const char*)(wh + (size_t)(n0 + row) * D_IN + k0) + c;
            asm volatile("global_load_async_to_lds_b128 %0, %1, off"
                         :: "v"(ldsa), "v"(ga) : "memory");
        }
    };

    // ---- fragment loads: 16-bit A/B striping (lane&15 = row, lane>>4 = K half)
    const int fr  = lane & 15;
    const int khB = (lane >> 4) * 32;     // byte offset of K-half

    auto loadFrag = [&](unsigned baseOff, int rbase) -> v16h {
        const char* p = lds + baseOff + (unsigned)(rbase + fr) * ROWB + khB;
        v8h lo = *(const v8h*)(p);
        v8h hi = *(const v8h*)(p + 16);
        return __builtin_shufflevector(lo, hi,
               0,1,2,3,4,5,6,7,8,9,10,11,12,13,14,15);
    };

    v8f acc[4][2] = {};

    auto compute = [&](int buf) {
        v16h a[4], b[2];
        #pragma unroll
        for (int mt = 0; mt < 4; ++mt)
            a[mt] = loadFrag(X_OFF + buf * TILEB, waveM + mt * 16);
        #pragma unroll
        for (int nt = 0; nt < 2; ++nt)
            b[nt] = loadFrag(W_OFF + buf * TILEB, waveN + nt * 16);
        #pragma unroll
        for (int mt = 0; mt < 4; ++mt) {
            #pragma unroll
            for (int nt = 0; nt < 2; ++nt) {
                acc[mt][nt] = __builtin_amdgcn_wmma_f32_16x16x32_f16(
                    false, a[mt], false, b[nt],
                    (short)0, acc[mt][nt], false, false);
            }
        }
    };

    // ---- pipeline: double-buffered LDS, one barrier per K-step ----
    loadRegsX(0);
    issueAsyncW(0, 0);
    storeLdsX(0);
    asm volatile("s_wait_asynccnt 0x0" ::: "memory");
    __syncthreads();

    for (int kt = 0; kt < KSTEPS; ++kt) {
        const int cur = kt & 1;
        const bool hasNext = (kt + 1) < KSTEPS;
        if (hasNext) {
            loadRegsX((kt + 1) * BK);
            issueAsyncW(cur ^ 1, (kt + 1) * BK);
        }
        compute(cur);
        if (hasNext) storeLdsX(cur ^ 1);
        asm volatile("s_wait_asynccnt 0x0" ::: "memory");
        __syncthreads();
    }

    // ---- epilogue: 16x16 f32 C/D layout -> global, fused bias ----
    #pragma unroll
    for (int nt = 0; nt < 2; ++nt) {
        const int n = n0 + waveN + nt * 16 + fr;
        const float bv = bias[n];
        #pragma unroll
        for (int mt = 0; mt < 4; ++mt) {
            #pragma unroll
            for (int v = 0; v < 8; ++v) {
                const int m = m0 + waveM + mt * 16 + (lane >> 4) * 8 + v;
                out[(size_t)m * D_OUT + n] = acc[mt][nt][v] + bv;
            }
        }
    }
}

extern "C" void kernel_launch(void* const* d_in, const int* in_sizes, int n_in,
                              void* d_out, int out_size, void* d_ws, size_t ws_size,
                              hipStream_t stream) {
    (void)in_sizes; (void)n_in; (void)out_size; (void)ws_size;
    const float* x      = (const float*)d_in[0];
    const int*   wq     = (const int*)d_in[1];
    const float* wscale = (const float*)d_in[2];
    const float* bias   = (const float*)d_in[3];
    float*       out    = (float*)d_out;
    _Float16*    wh     = (_Float16*)d_ws;      // 4096*4096*2 = 33.5 MB scratch

    // Phase 1: dequantize weights once (16.7M elems, 8 per thread).
    dequant_w_kernel<<<dim3((D_OUT * D_IN) / (256 * 8)), dim3(256), 0, stream>>>(
        wq, wscale, wh);

    // Phase 2: WMMA GEMM.
    dim3 grid(D_OUT / BN, M_TOT / BM);   // (32, 64)
    qlinear_wmma_kernel<<<grid, dim3(256), 0, stream>>>(x, wh, bias, out);
}
